// LSHAttention_4999341932659
// MI455X (gfx1250) — matrix operand order repaired and checked by reading, not compiled, not run on hardware
//
#include <hip/hip_runtime.h>
#include <hip/hip_bf16.h>

typedef _Float16 v16h __attribute__((ext_vector_type(16)));
typedef float    v8f  __attribute__((ext_vector_type(8)));

#define B_   2
#define H_   8
#define S_   2048
#define D_   64
#define NH   6
#define EPSF 1e-8f
#define KT   32          // keys per tile
#define QT   16          // queries per WMMA tile
#define UQ   2           // query tiles per wave
#define WAVES 4
#define NT   (S_ / KT)   // 64 key tiles

// 1/sqrt(D) * log2(e): WMMA produces scores directly in the log2 domain,
// so the masked softmax needs only a single v_exp_f32 (exp2) per element.
#define QSCALE (0.125f * 1.4426950408889634f)

#define WMMA_F16(A, Bm, C) \
    __builtin_amdgcn_wmma_f32_16x16x32_f16(false, (A), false, (Bm), (short)0, (C), false, false)

// ---------------------------------------------------------------------------
// Kernel 1: LSH bucket ids (sign of X . rotation; normalization is a positive
// scale and cannot change the sign bits).
// ---------------------------------------------------------------------------
__global__ void lsh_buckets_kernel(const float* __restrict__ Q,
                                   const float* __restrict__ K,
                                   const float* __restrict__ R,
                                   int* __restrict__ qbuck,
                                   int* __restrict__ kbuck) {
    const int total = B_ * H_ * S_;
    int idx = blockIdx.x * blockDim.x + threadIdx.x;   // [0, 2*total)
    const bool isK = idx >= total;
    int p = isK ? idx - total : idx;

    const float* X = (isK ? K : Q) + (size_t)p * D_;
    const int h = (p / S_) % H_;

    float x[D_];
    #pragma unroll
    for (int i = 0; i < D_; i += 4) {
        float4 t = *(const float4*)(X + i);
        x[i] = t.x; x[i + 1] = t.y; x[i + 2] = t.z; x[i + 3] = t.w;
    }

    int bucket = 0;
    #pragma unroll
    for (int n = 0; n < NH; ++n) {
        const float* r = R + ((size_t)h * NH + n) * D_;
        float acc = 0.f;
        #pragma unroll
        for (int i = 0; i < D_; ++i) acc += x[i] * r[i];
        bucket |= (acc > 0.f) ? (1 << n) : 0;
    }
    (isK ? kbuck : qbuck)[p] = bucket;
}

// ---------------------------------------------------------------------------
// Kernel 2: bucket-masked attention, max-free streaming softmax, WMMA f16.
// Block = 4 waves x 32 queries = 128 queries; double-buffered LDS staging.
// ---------------------------------------------------------------------------
__global__ __launch_bounds__(128) void lsh_attn_kernel(
        const float* __restrict__ Q, const float* __restrict__ K,
        const float* __restrict__ V, const int* __restrict__ qbuck,
        const int* __restrict__ kbuck, float* __restrict__ out) {

    __shared__ __align__(32) _Float16 Klds[2][KT * D_];        // [key][d]
    __shared__ __align__(32) _Float16 Vtlds[2][D_ * KT];       // [d][key]
    __shared__ __align__(32) _Float16 Pbuf[WAVES][UQ][QT * KT];
    __shared__ int kbLds[2][KT];

    const int tid  = threadIdx.x;
    const int wave = tid >> 5;
    const int lane = tid & 31;
    const int hi   = lane >> 4;       // 16-lane half selector
    const int ln   = lane & 15;

    const int qPerBlock    = WAVES * UQ * QT;                  // 128
    const int blocksPerSeq = S_ / qPerBlock;                   // 16
    const int bh = blockIdx.x / blocksPerSeq;
    const int qw = (blockIdx.x % blocksPerSeq) * qPerBlock + wave * (UQ * QT);

    const size_t base = (size_t)bh * S_ * D_;
    const float* Qg = Q + base;
    const float* Kg = K + base;
    const float* Vg = V + base;
    const int*   qb = qbuck + (size_t)bh * S_;
    const int*   kb = kbuck + (size_t)bh * S_;

    // --- Q tiles in 16-bit A layout: lane ln = row; K(e) = e + 16*hi
    v16h aq[UQ][2];
    #pragma unroll
    for (int u = 0; u < UQ; ++u) {
        const float* qrow = Qg + (size_t)(qw + u * QT + ln) * D_;
        #pragma unroll
        for (int e = 0; e < 16; ++e) {
            int d0 = e + 16 * hi;
            aq[u][0][e] = (_Float16)(qrow[d0]      * QSCALE);
            aq[u][1][e] = (_Float16)(qrow[d0 + 32] * QSCALE);
        }
    }

    int qbr[UQ][8];
    #pragma unroll
    for (int u = 0; u < UQ; ++u)
        #pragma unroll
        for (int r = 0; r < 8; ++r) qbr[u][r] = qb[qw + u * QT + r + 8 * hi];

    v8f   O[UQ][4];
    float lpart[UQ][8];
    #pragma unroll
    for (int u = 0; u < UQ; ++u) {
        #pragma unroll
        for (int t = 0; t < 4; ++t) O[u][t] = (v8f)0.f;
        #pragma unroll
        for (int r = 0; r < 8; ++r) lpart[u][r] = 0.f;
    }

    // --- staging registers (software pipeline) -----------------------------
    float4 kf[4], vf[4];
    int kbreg = 0;
    auto loadTile = [&](int kt) {
        const float4* Ksrc = (const float4*)(Kg + (size_t)kt * KT * D_);
        const float4* Vsrc = (const float4*)(Vg + (size_t)kt * KT * D_);
        #pragma unroll
        for (int j = 0; j < 4; ++j) {
            kf[j] = Ksrc[tid + j * 128];
            vf[j] = Vsrc[tid + j * 128];
        }
        if (tid < KT) kbreg = kb[kt * KT + tid];
    };
    auto storeTile = [&](int buf) {
        #pragma unroll
        for (int j = 0; j < 4; ++j) {
            const int c   = tid + j * 128;   // float4 chunk id
            const int key = c >> 4;          // (c*4)/64
            const int d   = (c & 15) * 4;
            _Float16* kd = &Klds[buf][key * D_ + d];
            kd[0] = (_Float16)kf[j].x; kd[1] = (_Float16)kf[j].y;
            kd[2] = (_Float16)kf[j].z; kd[3] = (_Float16)kf[j].w;
            _Float16* vt = &Vtlds[buf][d * KT + key];
            vt[0 * KT] = (_Float16)vf[j].x; vt[1 * KT] = (_Float16)vf[j].y;
            vt[2 * KT] = (_Float16)vf[j].z; vt[3 * KT] = (_Float16)vf[j].w;
        }
        if (tid < KT) kbLds[buf][tid] = kbreg;
    };

    loadTile(0);

    for (int kt = 0; kt < NT; ++kt) {
        const int buf = kt & 1;
        storeTile(buf);
        __syncthreads();

        const int nkt = (kt + 1 < NT) ? kt + 1 : NT - 1;
        loadTile(nkt);                       // overlaps with compute below

        const _Float16* KL = Klds[buf];
        const _Float16* VT = Vtlds[buf];
        const int kb0 = kbLds[buf][ln];
        const int kb1 = kbLds[buf][16 + ln];

        // --- S = Q K^T: shared B operands, 2 chained WMMAs per 16x16 tile --
        v16h bk[2][2];
        #pragma unroll
        for (int n = 0; n < 2; ++n) {
            const int keyl = n * 16 + ln;
            bk[n][0] = *(const v16h*)&KL[keyl * D_ + 16 * hi];       // d 0..31
            bk[n][1] = *(const v16h*)&KL[keyl * D_ + 32 + 16 * hi];  // d 32..63
        }
        v8f s[UQ][2];
        #pragma unroll
        for (int u = 0; u < UQ; ++u)
            #pragma unroll
            for (int n = 0; n < 2; ++n) {
                v8f c = (v8f)0.f;
                c = WMMA_F16(aq[u][0], bk[n][0], c);
                c = WMMA_F16(aq[u][1], bk[n][1], c);
                s[u][n] = c;
            }

        // --- max-free masked exp2 (scores already in log2 domain) ----------
        const float NEGINF = -__builtin_inff();
        #pragma unroll
        for (int u = 0; u < UQ; ++u) {
            _Float16* pb = Pbuf[wave][u];
            #pragma unroll
            for (int r = 0; r < 8; ++r) {
                float e0 = __builtin_amdgcn_exp2f((qbr[u][r] == kb0) ? s[u][0][r] : NEGINF);
                float e1 = __builtin_amdgcn_exp2f((qbr[u][r] == kb1) ? s[u][1][r] : NEGINF);
                lpart[u][r] += e0 + e1;
                const int row = r + 8 * hi;
                pb[row * KT + ln]      = (_Float16)e0;
                pb[row * KT + 16 + ln] = (_Float16)e1;
            }
        }

        // --- C layout -> A layout through wave-private LDS -----------------
        v16h ap[UQ];
        #pragma unroll
        for (int u = 0; u < UQ; ++u) {
            const _Float16* pb = Pbuf[wave][u];
            #pragma unroll
            for (int e = 0; e < 16; ++e) {
                int k0 = e + 16 * hi;
                ap[u][e] = pb[ln * KT + k0];
            }
        }

        // --- O += P V: shared B operands across query tiles ----------------
        #pragma unroll
        for (int t = 0; t < 4; ++t) {
            const v16h bv = *(const v16h*)&VT[(t * 16 + ln) * KT + 16 * hi];
            #pragma unroll
            for (int u = 0; u < UQ; ++u)
                O[u][t] = WMMA_F16(ap[u], bv, O[u][t]);
        }
    }

    // --- epilogue: single cross-lane reduction, out = O/(l+EPS)/NH ---------
    #pragma unroll
    for (int u = 0; u < UQ; ++u) {
        float scl[8];
        #pragma unroll
        for (int r = 0; r < 8; ++r) {
            float rs = lpart[u][r];
            #pragma unroll
            for (int off = 1; off < 16; off <<= 1)
                rs += __shfl_xor(rs, off, 32);
            scl[r] = __builtin_amdgcn_rcpf((rs + EPSF) * (float)NH);
        }
        #pragma unroll
        for (int t = 0; t < 4; ++t)
            #pragma unroll
            for (int r = 0; r < 8; ++r) {
                const int q = qw + u * QT + r + 8 * hi;
                out[base + (size_t)q * D_ + t * 16 + ln] = O[u][t][r] * scl[r];
            }
    }
}

// ---------------------------------------------------------------------------
extern "C" void kernel_launch(void* const* d_in, const int* in_sizes, int n_in,
                              void* d_out, int out_size, void* d_ws, size_t ws_size,
                              hipStream_t stream) {
    const float* Q = (const float*)d_in[0];
    const float* K = (const float*)d_in[1];
    const float* V = (const float*)d_in[2];
    const float* R = (const float*)d_in[3];
    float* out = (float*)d_out;

    int* qbuck = (int*)d_ws;
    int* kbuck = qbuck + B_ * H_ * S_;

    const int total2 = 2 * B_ * H_ * S_;                 // 65536
    lsh_buckets_kernel<<<total2 / 256, 256, 0, stream>>>(Q, K, R, qbuck, kbuck);

    const int nblocks = (B_ * H_ * S_) / (WAVES * UQ * QT);  // 256
    lsh_attn_kernel<<<nblocks, 128, 0, stream>>>(Q, K, V, qbuck, kbuck, out);
}
